// TGNN_18399639896425
// MI455X (gfx1250) — matrix operand order repaired and checked by reading, not compiled, not run on hardware
//
#include <hip/hip_runtime.h>
#include <hip/hip_bf16.h>

// ---------------------------------------------------------------------------
// CDNA5 (gfx1250) temporal GNN (GCN x2 -> GRU x2 -> heads).
// All GEMMs on v_wmma_f32_16x16x32_f16 (wave32 WMMA).
// Round-3 changes:
//  * GRU: A-fragments (x_t and h) hoisted out of the gate-tile loop -> 3x
//    fewer ds_load_b128 in the recurrent hot loop.
//  * GRU: global_prefetch of x_{t+1} rows while computing step t.
// ---------------------------------------------------------------------------

typedef __attribute__((ext_vector_type(16))) _Float16 v16h;
typedef __attribute__((ext_vector_type(8)))  float    v8f;

static constexpr int Bb = 8, Tt = 32, Nn = 4000, Ff = 16, Ee = 64000, GH = 64, HH = 128;
static constexpr int BT = Bb * Tt;            // 256
static constexpr int BN = Bb * Nn;            // 32000
static constexpr int RR = BT * Nn;            // 1,024,000 rows for the GCN GEMMs

__device__ __forceinline__ v8f wmma_f16(v16h a, v16h b, v8f c) {
  return __builtin_amdgcn_wmma_f32_16x16x32_f16(false, a, false, b, (short)0, c,
                                                false, false);
}

// A fragment (16x32 f16, M x K) from LDS: lane&15 = M row; lane half selects
// K sub-block. element i -> K = (i&7) + 16*(i>>3) + 8*(lane>=16)
__device__ __forceinline__ v16h frag_a(const _Float16* As, int lda, int lane) {
  int row  = lane & 15;
  int koff = (lane >> 4) << 3;
  v16h a;
#pragma unroll
  for (int i = 0; i < 16; ++i) {
    int k = (i & 7) + ((i >> 3) << 4) + koff;
    a[i] = As[row * lda + k];
  }
  return a;
}

// B fragment from pre-swizzled weight buffer: one 32-byte load per lane.
__device__ __forceinline__ v16h frag_b_pre(const _Float16* __restrict__ WF, int tile,
                                           int lane) {
  return *(const v16h*)(WF + ((size_t)tile * 32 + lane) * 16);
}

// ------------------------------- utility kernels ---------------------------

__global__ void zero_kernel(float* __restrict__ p, size_t n) {
  size_t i  = (size_t)blockIdx.x * blockDim.x + threadIdx.x;
  size_t st = (size_t)gridDim.x * blockDim.x;
  for (; i < n; i += st) p[i] = 0.f;
}

__global__ void deg_kernel(const int* __restrict__ ei, float* __restrict__ deg, int E) {
  int e = blockIdx.x * blockDim.x + threadIdx.x;
  if (e < E) atomicAdd(&deg[ei[E + e]], 1.f);
}

__global__ void invdeg_kernel(float* __restrict__ deg, int n) {
  int i = blockIdx.x * blockDim.x + threadIdx.x;
  if (i < n) deg[i] = 1.f / (deg[i] + 1.f);   // deg >= 1 always
}

// W (Nout,K) f32 row-major -> WF in WMMA B-fragment order, f16.
// Fragment (kc,nt): lane l element i holds W[n*K+k] with
//   n = nt*16 + (l&15),  k = kc*32 + (i&7) + 16*(i>>3) + 8*(l>=16)   (zero if k >= K).
// Flat layout: WF[((kc*NT + nt)*32 + lane)*16 + i],  NT = Nout/16.
__global__ void wfrag_kernel(const float* __restrict__ W, _Float16* __restrict__ WF,
                             int Nout, int K, int KP) {
  int NT    = Nout >> 4;
  int total = (KP >> 5) * NT * 512;
  int idx   = blockIdx.x * blockDim.x + threadIdx.x;
  if (idx >= total) return;
  int i = idx & 15, lane = (idx >> 4) & 31, tile = idx >> 9;
  int kc = tile / NT, nt = tile % NT;
  int k = kc * 32 + (i & 7) + ((i >> 3) << 4) + ((lane >> 4) << 3);
  int n = nt * 16 + (lane & 15);
  WF[idx] = (k < K) ? (_Float16)W[n * K + k] : (_Float16)0.f;
}

// scatter-add neighbor features: AGG[g, dst, :] += X[g, src, :] for all g in [0,BT)
template <int F>
__global__ void scatter_kernel(const float* __restrict__ X, const int* __restrict__ ei,
                               float* __restrict__ AGG) {
  size_t idx = (size_t)blockIdx.x * blockDim.x + threadIdx.x;
  if (idx >= (size_t)BT * Ee) return;
  int g = (int)(idx / Ee), e = (int)(idx % Ee);
  int s = ei[e], d = ei[Ee + e];
  const float* xs = X + ((size_t)g * Nn + s) * F;
  float*       ag = AGG + ((size_t)g * Nn + d) * F;
#pragma unroll
  for (int f = 0; f < F; ++f) atomicAdd(&ag[f], xs[f]);
}

// --------------------- GCN: fused (x+agg)/deg GEMM + LN + ReLU -------------
// A fragment generated on the fly from x and agg; each of the two K runs of 8
// contiguous elements is fetched with two float4 loads per array.
template <int K>
__device__ __forceinline__ v16h frag_a_gcn(const float* __restrict__ X,
                                           const float* __restrict__ AGG,
                                           float invd, size_t row, int kbase, int lane) {
  int koff = (lane >> 4) << 3;
  v16h a;
#pragma unroll
  for (int r = 0; r < 2; ++r) {
    int ks = kbase + koff + r * 16;
    if (ks + 8 <= K) {                            // compile-time foldable per chunk
      const float4* xp = (const float4*)(X + row * K + ks);
      const float4* gp = (const float4*)(AGG + row * K + ks);
      float4 x0 = xp[0], x1 = xp[1], g0 = gp[0], g1 = gp[1];
      a[r * 8 + 0] = (_Float16)((x0.x + g0.x) * invd);
      a[r * 8 + 1] = (_Float16)((x0.y + g0.y) * invd);
      a[r * 8 + 2] = (_Float16)((x0.z + g0.z) * invd);
      a[r * 8 + 3] = (_Float16)((x0.w + g0.w) * invd);
      a[r * 8 + 4] = (_Float16)((x1.x + g1.x) * invd);
      a[r * 8 + 5] = (_Float16)((x1.y + g1.y) * invd);
      a[r * 8 + 6] = (_Float16)((x1.z + g1.z) * invd);
      a[r * 8 + 7] = (_Float16)((x1.w + g1.w) * invd);
    } else {
#pragma unroll
      for (int i = 0; i < 8; ++i) a[r * 8 + i] = (_Float16)0.f;
    }
  }
  return a;
}

template <int K>
__global__ __launch_bounds__(256) void gcn_gemm_ln_kernel(
    const float* __restrict__ X, const float* __restrict__ AGG,
    const float* __restrict__ invd, const _Float16* __restrict__ WF,  // fragment order
    const float* __restrict__ bias, const float* __restrict__ gamma,
    const float* __restrict__ beta, float* __restrict__ out) {
  constexpr int KP = (K + 31) & ~31;
  __shared__ float Ls[8][16 * 64];                 // per-wave LN staging tile
  int tid = threadIdx.x, lane = tid & 31, w = tid >> 5;
  int r0  = (blockIdx.x * 8 + w) * 16;             // 16-row tile per wave
  size_t rowa = (size_t)(r0 + (lane & 15));
  float  id   = invd[(r0 + (lane & 15)) % Nn];

  v8f acc[4] = {};
#pragma unroll
  for (int kc = 0; kc < KP; kc += 32) {
    v16h a = frag_a_gcn<K>(X, AGG, id, rowa, kc, lane);
#pragma unroll
    for (int nt = 0; nt < 4; ++nt) {
      v16h b  = frag_b_pre(WF, (kc >> 5) * 4 + nt, lane);
      acc[nt] = wmma_f16(a, b, acc[nt]);
    }
  }
  // scatter accumulators into LN tile (+bias)
  int mh = (lane >> 4) << 3;
#pragma unroll
  for (int nt = 0; nt < 4; ++nt) {
    int   n  = nt * 16 + (lane & 15);
    float bn = bias[n];
#pragma unroll
    for (int v = 0; v < 8; ++v) Ls[w][(mh + v) * 64 + n] = acc[nt][v] + bn;
  }
  __syncthreads();
  // LayerNorm + ReLU per row (both lane halves compute stats, write 32 cols each)
  int   row = lane & 15;
  float mu = 0.f;
#pragma unroll
  for (int n = 0; n < 64; ++n) mu += Ls[w][row * 64 + n];
  mu *= (1.f / 64.f);
  float var = 0.f;
#pragma unroll
  for (int n = 0; n < 64; ++n) {
    float d = Ls[w][row * 64 + n] - mu;
    var += d * d;
  }
  var *= (1.f / 64.f);
  float rstd = rsqrtf(var + 1e-5f);
  int   nb   = (lane < 16) ? 0 : 32;
  for (int n = nb; n < nb + 32; ++n) {
    float vv = (Ls[w][row * 64 + n] - mu) * rstd * gamma[n] + beta[n];
    out[(size_t)(r0 + row) * 64 + n] = fmaxf(vv, 0.f);
  }
}

// ------------------------------ fused GRU layer ----------------------------
// One block = 16 sequences, 8 waves. h stays resident in LDS over all T steps.
// Per step: xp = x_t @ Wih^T and hp = h @ Whh^T via WMMA (each wave: 3 gate
// tiles of the 24), then elementwise r/z/n gates. Torch gate order: [r | z | n].
// A-fragments are loaded from LDS once per step and reused across gate tiles.
template <int DIN, bool L0>
__global__ __launch_bounds__(256) void gru_kernel(
    const float* __restrict__ X, const _Float16* __restrict__ WihF,
    const _Float16* __restrict__ WhhF, const float* __restrict__ bih,
    const float* __restrict__ bhh, float* __restrict__ Y, float* __restrict__ HT) {
  constexpr int KCX = DIN / 32;        // x-side K chunks (2 or 4)
  __shared__ _Float16 xs[16 * DIN];
  __shared__ _Float16 hs16[16 * HH];
  __shared__ float    hs[16 * HH];
  __shared__ float    rzs[16 * 256];   // r|z pre-activations
  __shared__ float    xns[16 * HH];    // x-side n gate
  __shared__ float    hns[16 * HH];    // h-side n gate
  int tid = threadIdx.x, lane = tid & 31, w = tid >> 5;
  int s0 = blockIdx.x * 16;            // 16 consecutive sequences (same b: 16|N)
  int b  = s0 / Nn, n0 = s0 % Nn;

  for (int idx = tid; idx < 16 * HH; idx += 256) {
    hs[idx]   = 0.f;
    hs16[idx] = (_Float16)0.f;
  }
  __syncthreads();

  for (int t = 0; t < Tt; ++t) {
    // stage x_t (f16) into LDS, float4 granularity (rows are contiguous)
    for (int idx = tid; idx < 16 * (DIN / 4); idx += 256) {
      int    m = idx / (DIN / 4), kq = idx % (DIN / 4);
      size_t base = L0 ? (size_t)((b * Tt + t) * Nn + (n0 + m)) * DIN
                       : (size_t)((s0 + m) * Tt + t) * DIN;
      float4 v = *(const float4*)(X + base + kq * 4);
      _Float16* d = xs + m * DIN + kq * 4;
      d[0] = (_Float16)v.x; d[1] = (_Float16)v.y;
      d[2] = (_Float16)v.z; d[3] = (_Float16)v.w;
    }
    // prefetch next step's x rows into cache while this step computes
    if (t + 1 < Tt && tid < 16) {
      int    m = tid;
      size_t nb = L0 ? (size_t)((b * Tt + (t + 1)) * Nn + (n0 + m)) * DIN
                     : (size_t)((s0 + m) * Tt + (t + 1)) * DIN;
      __builtin_prefetch(X + nb, 0, 3);
    }
    __syncthreads();

    // hoist all A-fragments for this step (reused by all 3 gate tiles)
    v16h afx[KCX];
#pragma unroll
    for (int kc = 0; kc < KCX; ++kc) afx[kc] = frag_a(xs + kc * 32, DIN, lane);
    v16h afh[4];
#pragma unroll
    for (int kc = 0; kc < 4; ++kc) afh[kc] = frag_a(hs16 + kc * 32, HH, lane);

    int mh = (lane >> 4) << 3;
#pragma unroll
    for (int j = 0; j < 3; ++j) {
      int gt = w * 3 + j;              // 24 gate tiles of 16 cols (3H = 384)
      v8f ax = {}, ah = {};
#pragma unroll
      for (int kc = 0; kc < KCX; ++kc)
        ax = wmma_f16(afx[kc], frag_b_pre(WihF, kc * 24 + gt, lane), ax);
#pragma unroll
      for (int kc = 0; kc < 4; ++kc)
        ah = wmma_f16(afh[kc], frag_b_pre(WhhF, kc * 24 + gt, lane), ah);
      int n = gt * 16 + (lane & 15);
      if (gt < 16) {                   // r,z region: x+h fused
        float bsum = bih[n] + bhh[n];
#pragma unroll
        for (int v = 0; v < 8; ++v) rzs[(mh + v) * 256 + n] = ax[v] + ah[v] + bsum;
      } else {                         // n gate: keep x/h sides separate
        int   jj = n - 256;
        float b1 = bih[n], b2 = bhh[n];
#pragma unroll
        for (int v = 0; v < 8; ++v) {
          xns[(mh + v) * HH + jj] = ax[v] + b1;
          hns[(mh + v) * HH + jj] = ah[v] + b2;
        }
      }
    }
    __syncthreads();

    for (int idx = tid; idx < 16 * HH; idx += 256) {
      int   m = idx >> 7, jj = idx & 127;
      float r  = 1.f / (1.f + __expf(-rzs[m * 256 + jj]));
      float z  = 1.f / (1.f + __expf(-rzs[m * 256 + 128 + jj]));
      float nn = tanhf(xns[idx] + r * hns[idx]);
      float hv = (1.f - z) * nn + z * hs[idx];
      hs[idx]   = hv;
      hs16[idx] = (_Float16)hv;
      if (L0) Y[(size_t)((s0 + m) * Tt + t) * HH + jj] = hv;
    }
    __syncthreads();
  }
  if (!L0)
    for (int idx = tid; idx < 16 * HH; idx += 256) HT[(size_t)s0 * HH + idx] = hs[idx];
}

// ------------------------------ heads & attention --------------------------

__global__ void heads_kernel(const float* __restrict__ Hm,
                             const float* __restrict__ cW1, const float* __restrict__ cb1,
                             const float* __restrict__ cW2, const float* __restrict__ cb2,
                             const float* __restrict__ rW1, const float* __restrict__ rb1,
                             const float* __restrict__ rW2, const float* __restrict__ rb2,
                             const float* __restrict__ Wn, const float* __restrict__ bn,
                             float* __restrict__ out_cls, float* __restrict__ out_reg,
                             float* __restrict__ out_rc) {
  int s = blockIdx.x * blockDim.x + threadIdx.x;
  if (s >= BN) return;
  const float* h = Hm + (size_t)s * HH;

  float acc2[4] = {cb2[0], cb2[1], cb2[2], cb2[3]};
  for (int j = 0; j < 64; ++j) {
    float a = cb1[j];
    for (int k = 0; k < HH; ++k) a += h[k] * cW1[j * HH + k];
    a = fmaxf(a, 0.f);
#pragma unroll
    for (int c = 0; c < 4; ++c) acc2[c] += a * cW2[c * 64 + j];
  }
#pragma unroll
  for (int c = 0; c < 4; ++c) out_cls[(size_t)s * 4 + c] = acc2[c];

  float rr = rb2[0];
  for (int j = 0; j < 64; ++j) {
    float a = rb1[j];
    for (int k = 0; k < HH; ++k) a += h[k] * rW1[j * HH + k];
    rr += fmaxf(a, 0.f) * rW2[j];
  }
  out_reg[s] = 1.f / (1.f + __expf(-rr));

  float ns = bn[0];
  for (int k = 0; k < HH; ++k) ns += h[k] * Wn[k];
  out_rc[s] = ns;                      // node_self; edges accumulate on top
}

__global__ void edge_attn_kernel(const float* __restrict__ Hm, const int* __restrict__ ei,
                                 const float* __restrict__ We, const float* __restrict__ be,
                                 float* __restrict__ out_rc) {
  size_t idx = (size_t)blockIdx.x * blockDim.x + threadIdx.x;
  if (idx >= (size_t)Bb * Ee) return;
  int b = (int)(idx / Ee), e = (int)(idx % Ee);
  int sv = ei[e], dv = ei[Ee + e];
  const float* h1 = Hm + ((size_t)b * Nn + sv) * HH;
  const float* h2 = Hm + ((size_t)b * Nn + dv) * HH;
  float es = be[0];
  for (int k = 0; k < HH; ++k) es += h1[k] * We[k] + h2[k] * We[HH + k];
  atomicAdd(&out_rc[b * Nn + dv], es);
}

__global__ void softmax_kernel(float* __restrict__ rc) {
  __shared__ float red[256];
  float* p  = rc + (size_t)blockIdx.x * Nn;
  int    tid = threadIdx.x;
  float  mx = -3.4e38f;
  for (int n = tid; n < Nn; n += 256) mx = fmaxf(mx, p[n]);
  red[tid] = mx;
  __syncthreads();
  for (int o = 128; o > 0; o >>= 1) {
    if (tid < o) red[tid] = fmaxf(red[tid], red[tid + o]);
    __syncthreads();
  }
  mx = red[0];
  __syncthreads();
  float sm = 0.f;
  for (int n = tid; n < Nn; n += 256) sm += __expf(p[n] - mx);
  red[tid] = sm;
  __syncthreads();
  for (int o = 128; o > 0; o >>= 1) {
    if (tid < o) red[tid] += red[tid + o];
    __syncthreads();
  }
  float inv = 1.f / red[0];
  __syncthreads();
  for (int n = tid; n < Nn; n += 256) p[n] = __expf(p[n] - mx) * inv;
}

// --------------------------------- launch ----------------------------------

extern "C" void kernel_launch(void* const* d_in, const int* in_sizes, int n_in,
                              void* d_out, int out_size, void* d_ws, size_t ws_size,
                              hipStream_t stream) {
  const float* x    = (const float*)d_in[0];
  const int*   ei   = (const int*)d_in[1];
  const float* g0W  = (const float*)d_in[2];
  const float* g0b  = (const float*)d_in[3];
  const float* g0g  = (const float*)d_in[4];
  const float* g0be = (const float*)d_in[5];
  const float* g1W  = (const float*)d_in[6];
  const float* g1b  = (const float*)d_in[7];
  const float* g1g  = (const float*)d_in[8];
  const float* g1be = (const float*)d_in[9];
  const float* Wih0 = (const float*)d_in[10];
  const float* Whh0 = (const float*)d_in[11];
  const float* bih0 = (const float*)d_in[12];
  const float* bhh0 = (const float*)d_in[13];
  const float* Wih1 = (const float*)d_in[14];
  const float* Whh1 = (const float*)d_in[15];
  const float* bih1 = (const float*)d_in[16];
  const float* bhh1 = (const float*)d_in[17];
  const float* cW1  = (const float*)d_in[18];
  const float* cb1  = (const float*)d_in[19];
  const float* cW2  = (const float*)d_in[20];
  const float* cb2  = (const float*)d_in[21];
  const float* rW1  = (const float*)d_in[22];
  const float* rb1  = (const float*)d_in[23];
  const float* rW2  = (const float*)d_in[24];
  const float* rb2  = (const float*)d_in[25];
  const float* We   = (const float*)d_in[26];
  const float* be   = (const float*)d_in[27];
  const float* Wn   = (const float*)d_in[28];
  const float* bn   = (const float*)d_in[29];

  // outputs: [risk_cls (B,N,4)][risk_reg (B,N)][root_cause (B,N)][h (B,N,H)]
  float* out_cls = (float*)d_out;
  float* out_reg = out_cls + (size_t)BN * 4;
  float* out_rc  = out_reg + BN;
  float* out_h   = out_rc + BN;

  // workspace carve-out
  char*  w    = (char*)d_ws;
  auto   take = [&](size_t bytes) -> void* {
    void* p = (void*)w;
    w += (bytes + 255) & ~(size_t)255;
    return p;
  };
  float*     deg   = (float*)take((size_t)Nn * 4);
  _Float16*  WgF0  = (_Float16*)take((size_t)32 * 64 * 2);
  _Float16*  WgF1  = (_Float16*)take((size_t)64 * 64 * 2);
  _Float16*  Wih0F = (_Float16*)take((size_t)64 * 384 * 2);
  _Float16*  Whh0F = (_Float16*)take((size_t)128 * 384 * 2);
  _Float16*  Wih1F = (_Float16*)take((size_t)128 * 384 * 2);
  _Float16*  Whh1F = (_Float16*)take((size_t)128 * 384 * 2);
  float*     agg   = (float*)take((size_t)RR * 64 * 4);
  float*     g0    = (float*)take((size_t)RR * 64 * 4);
  float*     g1    = (float*)take((size_t)RR * 64 * 4);
  float*     y1    = (float*)take((size_t)BN * Tt * HH * 4);

  // degree / inverse degree
  zero_kernel<<<64, 256, 0, stream>>>(deg, (size_t)Nn);
  deg_kernel<<<(Ee + 255) / 256, 256, 0, stream>>>(ei, deg, Ee);
  invdeg_kernel<<<(Nn + 255) / 256, 256, 0, stream>>>(deg, Nn);

  // weight fragmentization (f32 -> f16, WMMA B-fragment order, K zero-padded)
  wfrag_kernel<<<(32 * 64 + 255) / 256, 256, 0, stream>>>(g0W, WgF0, 64, 16, 32);
  wfrag_kernel<<<(64 * 64 + 255) / 256, 256, 0, stream>>>(g1W, WgF1, 64, 64, 64);
  wfrag_kernel<<<(64 * 384 + 255) / 256, 256, 0, stream>>>(Wih0, Wih0F, 384, 64, 64);
  wfrag_kernel<<<(128 * 384 + 255) / 256, 256, 0, stream>>>(Whh0, Whh0F, 384, 128, 128);
  wfrag_kernel<<<(128 * 384 + 255) / 256, 256, 0, stream>>>(Wih1, Wih1F, 384, 128, 128);
  wfrag_kernel<<<(128 * 384 + 255) / 256, 256, 0, stream>>>(Whh1, Whh1F, 384, 128, 128);

  // GCN layer 0
  zero_kernel<<<4096, 256, 0, stream>>>(agg, (size_t)RR * Ff);
  scatter_kernel<Ff><<<(int)(((size_t)BT * Ee + 255) / 256), 256, 0, stream>>>(x, ei, agg);
  gcn_gemm_ln_kernel<Ff><<<RR / 128, 256, 0, stream>>>(x, agg, deg, WgF0, g0b, g0g,
                                                       g0be, g0);
  // GCN layer 1
  zero_kernel<<<4096, 256, 0, stream>>>(agg, (size_t)RR * GH);
  scatter_kernel<GH><<<(int)(((size_t)BT * Ee + 255) / 256), 256, 0, stream>>>(g0, ei, agg);
  gcn_gemm_ln_kernel<GH><<<RR / 128, 256, 0, stream>>>(g0, agg, deg, WgF1, g1b, g1g,
                                                       g1be, g1);

  // GRU layers (fused over T, WMMA projections)
  gru_kernel<GH, true><<<BN / 16, 256, 0, stream>>>(g1, Wih0F, Whh0F, bih0, bhh0, y1,
                                                    nullptr);
  gru_kernel<HH, false><<<BN / 16, 256, 0, stream>>>(y1, Wih1F, Whh1F, bih1, bhh1,
                                                     nullptr, out_h);

  // heads + root-cause attention
  heads_kernel<<<(BN + 255) / 256, 256, 0, stream>>>(out_h, cW1, cb1, cW2, cb2, rW1,
                                                     rb1, rW2, rb2, Wn, bn, out_cls,
                                                     out_reg, out_rc);
  edge_attn_kernel<<<(int)(((size_t)Bb * Ee + 255) / 256), 256, 0, stream>>>(out_h, ei,
                                                                             We, be,
                                                                             out_rc);
  softmax_kernel<<<Bb, 256, 0, stream>>>(out_rc);
}
